// S6_51367808860423
// MI455X (gfx1250) — matrix operand order repaired and checked by reading, compiled, not run on hardware
//
#include <hip/hip_runtime.h>
#include <hip/hip_bf16.h>
#include <math.h>

#define D_MODEL 1024
#define D_STATE 16
#define D_CONV  4
#define DT_RANK 64
#define BATCH   4
#define SEQLEN  1024
#define NPROJ   (DT_RANK + 2 * D_STATE)   // 96
#define M_TOT   (BATCH * SEQLEN)          // 4096
#define CHUNK   8

typedef __bf16 bf16_t;
typedef __attribute__((ext_vector_type(16))) __bf16 v16bf;
typedef __attribute__((ext_vector_type(8)))  __bf16 v8bf;
typedef __attribute__((ext_vector_type(8)))  float  v8f;

__device__ __forceinline__ float silu_f(float v) {
    return v / (1.0f + __expf(-v));
}

// ---------------------------------------------------------------------------
// Kernel 0: convert GEMM weights f32 -> bf16 (both stay L2-resident).
// ---------------------------------------------------------------------------
__global__ void k_cvt_weights(const float* __restrict__ w1, const float* __restrict__ wd,
                              bf16_t* __restrict__ w1b, bf16_t* __restrict__ wdb) {
    int i = blockIdx.x * 256 + threadIdx.x;
    if (i < NPROJ * D_MODEL)  w1b[i] = (bf16_t)w1[i];
    if (i < D_MODEL * DT_RANK) wdb[i] = (bf16_t)wd[i];
}

// ---------------------------------------------------------------------------
// Kernel 1: depthwise causal conv (width 4, left pad 3) + bias + SiLU.
// Writes f32 activation (for scan) and bf16 copy (WMMA A operand).
// Fully coalesced: consecutive threads -> consecutive d.
// ---------------------------------------------------------------------------
__global__ void k_conv_silu(const float* __restrict__ mod, const float* __restrict__ cw,
                            const float* __restrict__ cb,
                            float* __restrict__ xact, bf16_t* __restrict__ xb) {
    int idx = blockIdx.x * 256 + threadIdx.x;           // over B*L*D
    int d = idx % D_MODEL;
    int l = (idx / D_MODEL) % SEQLEN;
    int b = idx / (D_MODEL * SEQLEN);
    float acc = cb[d];
#pragma unroll
    for (int k = 0; k < D_CONV; ++k) {
        int ll = l - (D_CONV - 1) + k;
        if (ll >= 0)
            acc += mod[((size_t)b * SEQLEN + ll) * (2 * D_MODEL) + d] * cw[d * D_CONV + k];
    }
    float s = silu_f(acc);
    xact[idx] = s;
    xb[idx]   = (bf16_t)s;
}

// ---------------------------------------------------------------------------
// Kernel 2: GEMM1  x_dbl[m, r] = sum_k x[m,k] * W1[r,k]
// M=4096, K=1024, N=96. One wave per 16x16 tile, v_wmma_f32_16x16x32_bf16.
// Per-lane fragment loads are contiguous 16B/32B runs per the CDNA5 layouts.
// Also emits bf16 copy of the dt slice (r < 64) as GEMM2's A operand.
// ---------------------------------------------------------------------------
__global__ void k_gemm_xdbl(const bf16_t* __restrict__ X, const bf16_t* __restrict__ W,
                            float* __restrict__ xdbl, bf16_t* __restrict__ dtb) {
    const int lane = threadIdx.x & 31;
    const int tile = blockIdx.x * (blockDim.x >> 5) + (threadIdx.x >> 5);
    const int mt = tile / (NPROJ / 16);
    const int nt = tile % (NPROJ / 16);
    const int m0 = mt * 16, n0 = nt * 16;
    const int mrow = m0 + (lane & 15);
    const int nrow = n0 + (lane & 15);
    const int hiA = (lane >> 4) * 8;    // A k-base: lanes 16-31 hold K {8..15},{24..31}
    const int hiB = (lane >> 4) * 8;    // B k-base: lanes 16-31 hold K {8..15},{24..31}
    v8f acc = {};
    for (int k0 = 0; k0 < D_MODEL; k0 += 32) {
        v8bf a_lo = *(const v8bf*)(X + (size_t)mrow * D_MODEL + k0 + hiA);
        v8bf a_hi = *(const v8bf*)(X + (size_t)mrow * D_MODEL + k0 + hiA + 16);
        v16bf a;
#pragma unroll
        for (int e = 0; e < 8; ++e) { a[e] = a_lo[e]; a[e + 8] = a_hi[e]; }
        v8bf b_lo = *(const v8bf*)(W + (size_t)nrow * D_MODEL + k0 + hiB);
        v8bf b_hi = *(const v8bf*)(W + (size_t)nrow * D_MODEL + k0 + hiB + 16);
        v16bf bf;
#pragma unroll
        for (int e = 0; e < 8; ++e) { bf[e] = b_lo[e]; bf[e + 8] = b_hi[e]; }
        acc = __builtin_amdgcn_wmma_f32_16x16x32_bf16(false, a, false, bf,
                                                      (short)0, acc, false, false);
    }
    const int moff = m0 + ((lane >> 4) * 8);
#pragma unroll
    for (int j = 0; j < 8; ++j) {
        int m = moff + j;
        float v = acc[j];
        xdbl[(size_t)m * NPROJ + nrow] = v;
        if (nrow < DT_RANK) dtb[(size_t)m * DT_RANK + nrow] = (bf16_t)v;
    }
}

// ---------------------------------------------------------------------------
// Kernel 3: GEMM2  delta[m, d] = softplus( sum_r dt[m,r]*Wd[d,r] + bias[d] )
// M=4096, K=64, N=1024. Same wave-per-tile WMMA scheme, fused epilogue.
// ---------------------------------------------------------------------------
__global__ void k_gemm_delta(const bf16_t* __restrict__ DT, const bf16_t* __restrict__ WD,
                             const float* __restrict__ bias, float* __restrict__ delta) {
    const int lane = threadIdx.x & 31;
    const int tile = blockIdx.x * (blockDim.x >> 5) + (threadIdx.x >> 5);
    const int mt = tile / (D_MODEL / 16);
    const int nt = tile % (D_MODEL / 16);
    const int m0 = mt * 16, n0 = nt * 16;
    const int mrow = m0 + (lane & 15);
    const int nrow = n0 + (lane & 15);
    const int hiA = (lane >> 4) * 8;
    const int hiB = (lane >> 4) * 8;
    v8f acc = {};
#pragma unroll
    for (int k0 = 0; k0 < DT_RANK; k0 += 32) {
        v8bf a_lo = *(const v8bf*)(DT + (size_t)mrow * DT_RANK + k0 + hiA);
        v8bf a_hi = *(const v8bf*)(DT + (size_t)mrow * DT_RANK + k0 + hiA + 16);
        v16bf a;
#pragma unroll
        for (int e = 0; e < 8; ++e) { a[e] = a_lo[e]; a[e + 8] = a_hi[e]; }
        v8bf b_lo = *(const v8bf*)(WD + (size_t)nrow * DT_RANK + k0 + hiB);
        v8bf b_hi = *(const v8bf*)(WD + (size_t)nrow * DT_RANK + k0 + hiB + 16);
        v16bf bf;
#pragma unroll
        for (int e = 0; e < 8; ++e) { bf[e] = b_lo[e]; bf[e + 8] = b_hi[e]; }
        acc = __builtin_amdgcn_wmma_f32_16x16x32_bf16(false, a, false, bf,
                                                      (short)0, acc, false, false);
    }
    const float bv = bias[nrow];
    const int moff = m0 + ((lane >> 4) * 8);
#pragma unroll
    for (int j = 0; j < 8; ++j) {
        float v = acc[j] + bv;
        float sp = (v > 20.0f) ? v : log1pf(__expf(v));   // softplus
        delta[(size_t)(moff + j) * D_MODEL + nrow] = sp;
    }
}

// ---------------------------------------------------------------------------
// Kernel 4: selective scan. One thread per (b,d) chain, h[16]+A[16] in VGPRs.
// Lane == consecutive d -> coalesced delta/x/z loads; per-step B/C rows (32
// floats shared by the whole block) staged in LDS in CHUNK-step batches.
// Writes y to (B,L,D) coalesced; transpose happens in kernel 5.
// ---------------------------------------------------------------------------
__global__ void k_scan(const float* __restrict__ delta, const float* __restrict__ xact,
                       const float* __restrict__ xdbl, const float* __restrict__ Alog,
                       const float* __restrict__ Dp, const float* __restrict__ mod,
                       float* __restrict__ ybuf) {
    __shared__ float sBC[CHUNK][2 * D_STATE];
    const int b = blockIdx.y;
    const int d = blockIdx.x * 256 + threadIdx.x;
    float An[D_STATE];
#pragma unroll
    for (int n = 0; n < D_STATE; ++n) An[n] = -__expf(Alog[d * D_STATE + n]);
    float h[D_STATE] = {};
    const float Dv = Dp[d];
    const size_t baseBL = (size_t)b * SEQLEN;
    for (int l0 = 0; l0 < SEQLEN; l0 += CHUNK) {
        {   // 256 threads cooperatively fetch CHUNK*32 B/C values
            int step = threadIdx.x >> 5;
            int e    = threadIdx.x & 31;
            sBC[step][e] = xdbl[(baseBL + l0 + step) * NPROJ + DT_RANK + e];
        }
        __syncthreads();
#pragma unroll
        for (int s = 0; s < CHUNK; ++s) {
            size_t idx = (baseBL + l0 + s) * D_MODEL + d;
            float dl = delta[idx];
            float xv = xact[idx];
            float y = 0.0f;
#pragma unroll
            for (int n = 0; n < D_STATE; ++n) {
                float dA  = __expf(dl * An[n]);
                float dBu = dl * sBC[s][n] * xv;
                h[n] = dA * h[n] + dBu;
                y += h[n] * sBC[s][D_STATE + n];
            }
            y += xv * Dv;
            float zv = mod[(baseBL + l0 + s) * (2 * D_MODEL) + D_MODEL + d];
            y *= silu_f(zv);
            ybuf[idx] = y;
        }
        __syncthreads();
    }
}

// ---------------------------------------------------------------------------
// Kernel 5: (B,L,D) -> (B,D,L) transpose via 32x33 LDS tiles (coalesced both
// ways; padding kills the 32-way bank conflict).
// ---------------------------------------------------------------------------
__global__ void k_transpose(const float* __restrict__ ybuf, float* __restrict__ out) {
    __shared__ float tile[32][33];
    const int b  = blockIdx.z;
    const int d0 = blockIdx.x * 32;
    const int l0 = blockIdx.y * 32;
    const float* src = ybuf + (size_t)b * SEQLEN * D_MODEL;
    float*       dst = out  + (size_t)b * D_MODEL * SEQLEN;
    for (int r = threadIdx.y; r < 32; r += 8)
        tile[r][threadIdx.x] = src[(size_t)(l0 + r) * D_MODEL + d0 + threadIdx.x];
    __syncthreads();
    for (int r = threadIdx.y; r < 32; r += 8)
        dst[(size_t)(d0 + r) * SEQLEN + l0 + threadIdx.x] = tile[threadIdx.x][r];
}

// ---------------------------------------------------------------------------
extern "C" void kernel_launch(void* const* d_in, const int* in_sizes, int n_in,
                              void* d_out, int out_size, void* d_ws, size_t ws_size,
                              hipStream_t stream) {
    const float* mod    = (const float*)d_in[0];   // (B,L,2D)
    const float* conv_w = (const float*)d_in[1];   // (D,1,4)
    const float* conv_b = (const float*)d_in[2];   // (D,)
    const float* xpw    = (const float*)d_in[3];   // (96,D)
    const float* dtw    = (const float*)d_in[4];   // (D,64)
    const float* dtb_b  = (const float*)d_in[5];   // (D,)
    const float* Alog   = (const float*)d_in[6];   // (D,16)
    const float* Dp     = (const float*)d_in[7];   // (D,)
    float* out = (float*)d_out;                    // (B,D,L)

    char* ws = (char*)d_ws;
    // byte offsets (all 4KB aligned)
    float*  xact  = (float*) (ws + 0);                       // 16 MB
    bf16_t* xb    = (bf16_t*)(ws + 16777216);                //  8 MB
    float*  xdbl  = (float*) (ws + 25165824);                //  1.5 MB
    bf16_t* dtb   = (bf16_t*)(ws + 26738688);                //  0.5 MB
    float*  delta = (float*) (ws + 27262976);                // 16 MB
    float*  ybuf  = (float*) (ws + 44040192);                // 16 MB
    bf16_t* w1b   = (bf16_t*)(ws + 60817408);                // 192 KB
    bf16_t* wdb   = (bf16_t*)(ws + 61014016);                // 128 KB

    // 0: weight conversion
    k_cvt_weights<<<(NPROJ * D_MODEL + 255) / 256, 256, 0, stream>>>(xpw, dtw, w1b, wdb);

    // 1: conv + SiLU
    k_conv_silu<<<(BATCH * SEQLEN * D_MODEL) / 256, 256, 0, stream>>>(
        mod, conv_w, conv_b, xact, xb);

    // 2: GEMM1 (M=4096, N=96) -> 1536 wave-tiles, 8 waves per block
    k_gemm_xdbl<<<(M_TOT / 16) * (NPROJ / 16) / 8, 256, 0, stream>>>(xb, w1b, xdbl, dtb);

    // 3: GEMM2 (M=4096, N=1024) -> 16384 wave-tiles
    k_gemm_delta<<<(M_TOT / 16) * (D_MODEL / 16) / 8, 256, 0, stream>>>(
        dtb, wdb, dtb_b, delta);

    // 4: selective scan: grid (D/256, B)
    k_scan<<<dim3(D_MODEL / 256, BATCH), 256, 0, stream>>>(
        delta, xact, xdbl, Alog, Dp, mod, ybuf);

    // 5: transpose to (B,D,L)
    k_transpose<<<dim3(D_MODEL / 32, SEQLEN / 32, BATCH), dim3(32, 8), 0, stream>>>(
        ybuf, out);
}
